// cross_att_45380624449881
// MI455X (gfx1250) — matrix operand order repaired and checked by reading, not compiled, not run on hardware
//
#include <hip/hip_runtime.h>
#include <hip/hip_bf16.h>

typedef __attribute__((ext_vector_type(16))) _Float16 v16h;
typedef __attribute__((ext_vector_type(8)))  _Float16 v8h;
typedef __attribute__((ext_vector_type(4)))  _Float16 v4h;
typedef __attribute__((ext_vector_type(8)))  float    v8f;
typedef __attribute__((ext_vector_type(4)))  float    v4f;

#define B_SZ      16
#define N_SZ      2048
#define D_SZ      512
#define M_Q       64          // query rows per workgroup (4 subtiles of 16)
#define BLK_K     64          // keys per main-loop iteration
#define NWAVES    16
#define NTHREADS  512
#define D_SLICE   128         // output D columns per wave (4*128 = 512)
#define NORM_FACT 0.044194173824159216f   // 1/sqrt(512)

// padded LDS strides (halves); multiples of 8 halves keep 16B alignment
#define QS_STRIDE 520
#define KS_STRIDE 520
#define VT_STRIDE (BLK_K + 8)   // 72
#define PS_STRIDE (BLK_K + 8)   // 72

#define QS_ELEMS (M_Q   * QS_STRIDE)   // 33280
#define KS_ELEMS (BLK_K * KS_STRIDE)   // 33280
#define VT_ELEMS (D_SZ  * VT_STRIDE)   // 36864
#define PS_ELEMS (M_Q   * PS_STRIDE)   // 4608
#define SMEM_HALVES (QS_ELEMS + KS_ELEMS + VT_ELEMS + PS_ELEMS)
#define SMEM_BYTES  (SMEM_HALVES * 2 + (NWAVES*16 + NWAVES*16 + M_Q + M_Q) * 4)

__device__ __forceinline__ v16h make_frag(v8h lo, v8h hi) {
  v16h r;
#pragma unroll
  for (int j = 0; j < 8; ++j) { r[j] = lo[j]; r[j + 8] = hi[j]; }
  return r;
}

// issue one 16B async DMA copy: global (saddr + voff) -> LDS byte address
__device__ __forceinline__ void async_b128(unsigned lds_byte, unsigned goff,
                                           const _Float16* sbase) {
  asm volatile("global_load_async_to_lds_b128 %0, %1, %2"
               :: "v"(lds_byte), "v"(goff), "s"(sbase) : "memory");
}
__device__ __forceinline__ void wait_async0() {
  asm volatile("s_wait_asynccnt 0" ::: "memory");
}

// ---------------- pre-pass kernels ----------------
__global__ __launch_bounds__(256)
void cvt_f16_kernel(const float* __restrict__ in, _Float16* __restrict__ outp, int n4) {
  int i = blockIdx.x * 256 + threadIdx.x;
  if (i < n4) {
    v4f x = ((const v4f*)in)[i];
    v4h h; h[0] = (_Float16)x[0]; h[1] = (_Float16)x[1];
           h[2] = (_Float16)x[2]; h[3] = (_Float16)x[3];
    ((v4h*)outp)[i] = h;
  }
}

// V[b][n][d] (f32) -> Vt[b][d][n] (f16), 32x32 LDS tiles
__global__ __launch_bounds__(256)
void transpose_f16_kernel(const float* __restrict__ V, _Float16* __restrict__ Vt) {
  __shared__ float tile[32][33];
  int b  = blockIdx.z;
  int n0 = blockIdx.x * 32;
  int d0 = blockIdx.y * 32;
  const float* src = V + ((size_t)b * N_SZ + n0) * D_SZ + d0;
#pragma unroll
  for (int r = threadIdx.y; r < 32; r += 8)
    tile[r][threadIdx.x] = src[(size_t)r * D_SZ + threadIdx.x];
  __syncthreads();
  _Float16* dst = Vt + ((size_t)b * D_SZ + d0) * N_SZ + n0;
#pragma unroll
  for (int r = threadIdx.y; r < 32; r += 8)
    dst[(size_t)r * N_SZ + threadIdx.x] = (_Float16)tile[threadIdx.x][r];
}

// ---------------- main flash-attention kernel ----------------
template <bool PRECONV>
__global__ __launch_bounds__(NTHREADS)
void flash_cross_attn(const float* __restrict__ key,
                      const float* __restrict__ query,
                      const float* __restrict__ value,
                      const _Float16* __restrict__ Kh,   // [B][N][D] f16 (PRECONV)
                      const _Float16* __restrict__ Vth,  // [B][D][N] f16 (PRECONV)
                      float* __restrict__ out) {
  extern __shared__ _Float16 smem[];
  _Float16* Qs = smem;                   // [64][520]
  _Float16* Ks = Qs + QS_ELEMS;          // [64][520]
  _Float16* Vt = Ks + KS_ELEMS;          // [512][72] transposed
  _Float16* Ps = Vt + VT_ELEMS;          // [64][72]
  float* red    = (float*)(Ps + PS_ELEMS);   // [4 qi][4 di][16]
  float* sumred = red + NWAVES * 16;
  float* rowm   = sumred + NWAVES * 16;      // [64]
  float* rowl   = rowm + M_Q;                // [64]

  const int tid  = threadIdx.x;
  const int lane = tid & 31;
  const int wave = tid >> 5;
  const int qi   = wave >> 2;       // q-subtile 0..3
  const int di   = wave & 3;        // d-slice / key-column group 0..3
  const int lh   = lane >> 4;
  const int rA   = lane & 15;

  const int b  = blockIdx.y;
  const int q0 = blockIdx.x * M_Q;
  const float* kbase = key   + (size_t)b * N_SZ * D_SZ;
  const float* qbase = query + (size_t)b * N_SZ * D_SZ;
  const float* vbase = value + (size_t)b * N_SZ * D_SZ;
  float*       obase = out   + (size_t)b * N_SZ * D_SZ;

  // ---- stage Q tile (f32 -> f16), init softmax state ----
  for (int i = tid; i < M_Q * (D_SZ / 4); i += NTHREADS) {
    int row = i >> 7;
    int c4  = (i & 127) * 4;
    v4f q = *(const v4f*)(qbase + (size_t)(q0 + row) * D_SZ + c4);
    v4h h; h[0] = (_Float16)q[0]; h[1] = (_Float16)q[1];
           h[2] = (_Float16)q[2]; h[3] = (_Float16)q[3];
    *(v4h*)(Qs + row * QS_STRIDE + c4) = h;
  }
  if (tid < M_Q) { rowm[tid] = -__builtin_inff(); rowl[tid] = 0.0f; }

  v8f o[8];
#pragma unroll
  for (int t = 0; t < 8; ++t) { v8f z = {}; o[t] = z; }

  const _Float16* qrowp = Qs + (qi * 16 + rA) * QS_STRIDE + lh * 8;
  const _Float16* krowp = Ks + (di * 16 + rA) * KS_STRIDE + lh * 8;
  const _Float16* prowp = Ps + (qi * 16 + rA) * PS_STRIDE + lh * 8;

  const unsigned ksB = (unsigned)(uintptr_t)Ks;   // LDS byte offsets
  const unsigned vtB = (unsigned)(uintptr_t)Vt;

  for (int kb = 0; kb < N_SZ; kb += BLK_K) {
    // ---- stage K tile (row-major) and Vt tile (d-major) ----
    if (PRECONV) {
      const _Float16* khg = Kh  + ((size_t)b * N_SZ + kb) * D_SZ;
      const _Float16* vtg = Vth + (size_t)b * D_SZ * N_SZ + kb;
#pragma unroll
      for (int j = 0; j < 8; ++j) {                   // 4096 16B chunks each
        int c = tid + j * NTHREADS;
        int krow = c >> 6, koff = (c & 63) * 16;      // 64 chunks per K row
        async_b128(ksB + krow * (KS_STRIDE * 2) + koff,
                   (unsigned)(krow * (D_SZ * 2) + koff), khg);
        int dr = c >> 3, voff = (c & 7) * 16;         // 8 chunks per Vt row
        async_b128(vtB + dr * (VT_STRIDE * 2) + voff,
                   (unsigned)(dr * (N_SZ * 2) + voff), vtg);
        if ((c & 31) == 0) {  // speculative prefetch of next key-block into L2
          __builtin_prefetch((const char*)khg + (size_t)BLK_K * D_SZ * 2 +
                             krow * (D_SZ * 2) + koff, 0, 0);
        }
      }
      wait_async0();
    } else {
      for (int i = tid; i < BLK_K * (D_SZ / 4); i += NTHREADS) {
        int row = i >> 7;
        int c4  = (i & 127) * 4;
        v4f kv = *(const v4f*)(kbase + (size_t)(kb + row) * D_SZ + c4);
        v4f vv = *(const v4f*)(vbase + (size_t)(kb + row) * D_SZ + c4);
        v4h kh; kh[0] = (_Float16)kv[0]; kh[1] = (_Float16)kv[1];
                kh[2] = (_Float16)kv[2]; kh[3] = (_Float16)kv[3];
        *(v4h*)(Ks + row * KS_STRIDE + c4) = kh;
        Vt[(c4 + 0) * VT_STRIDE + row] = (_Float16)vv[0];
        Vt[(c4 + 1) * VT_STRIDE + row] = (_Float16)vv[1];
        Vt[(c4 + 2) * VT_STRIDE + row] = (_Float16)vv[2];
        Vt[(c4 + 3) * VT_STRIDE + row] = (_Float16)vv[3];
      }
    }
    __syncthreads();   // (A) tiles visible

    // ---- GEMM1: S(16x16) = Q[qi] . K^T[cols 16*di..]  over D=512 ----
    v8f s = {};
#pragma unroll 4
    for (int kd = 0; kd < D_SZ; kd += 32) {
      v16h a  = make_frag(*(const v8h*)(qrowp + kd), *(const v8h*)(qrowp + kd + 16));
      v16h bb = make_frag(*(const v8h*)(krowp + kd), *(const v8h*)(krowp + kd + 16));
      s = __builtin_amdgcn_wmma_f32_16x16x32_f16(false, a, false, bb,
                                                 (short)0, s, false, false);
    }

    // ---- online softmax ----
    float mloc[8];
#pragma unroll
    for (int v = 0; v < 8; ++v) mloc[v] = s[v];
#pragma unroll
    for (int off = 8; off >= 1; off >>= 1)
#pragma unroll
      for (int v = 0; v < 8; ++v)
        mloc[v] = fmaxf(mloc[v], __shfl_xor(mloc[v], off, 32));
    if (rA == 0) {
#pragma unroll
      for (int v = 0; v < 8; ++v) red[wave * 16 + v + 8 * lh] = mloc[v];
    }
    __syncthreads();   // (B1)

    float mnew[8], alpha[8], p[8];
#pragma unroll
    for (int v = 0; v < 8; ++v) {
      int r16 = v + 8 * lh;
      float mo = rowm[qi * 16 + r16];
      float mb = red[(qi * 4 + 0) * 16 + r16];
#pragma unroll
      for (int w = 1; w < 4; ++w) mb = fmaxf(mb, red[(qi * 4 + w) * 16 + r16]);
      float mn = fmaxf(mo, mb);
      mnew[v]  = mn;
      alpha[v] = __expf(mo - mn);
      p[v]     = __expf(s[v] - mn);
    }
#pragma unroll
    for (int t = 0; t < 8; ++t)
#pragma unroll
      for (int v = 0; v < 8; ++v) o[t][v] *= alpha[v];

    float sl[8];
#pragma unroll
    for (int v = 0; v < 8; ++v) sl[v] = p[v];
#pragma unroll
    for (int off = 8; off >= 1; off >>= 1)
#pragma unroll
      for (int v = 0; v < 8; ++v) sl[v] += __shfl_xor(sl[v], off, 32);
    if (rA == 0) {
#pragma unroll
      for (int v = 0; v < 8; ++v) sumred[wave * 16 + v + 8 * lh] = sl[v];
    }
#pragma unroll
    for (int v = 0; v < 8; ++v)
      Ps[(qi * 16 + v + 8 * lh) * PS_STRIDE + di * 16 + rA] = (_Float16)p[v];
    __syncthreads();   // (B2)

    if (di == 0 && rA == 0) {   // one wave per q-subtile updates running stats
#pragma unroll
      for (int v = 0; v < 8; ++v) {
        int r16 = v + 8 * lh, grow = qi * 16 + r16;
        float tot = 0.0f;
#pragma unroll
        for (int w = 0; w < 4; ++w) tot += sumred[(qi * 4 + w) * 16 + r16];
        rowl[grow] = rowl[grow] * alpha[v] + tot;
        rowm[grow] = mnew[v];
      }
    }
    __syncthreads();   // (C)

    // ---- GEMM2: O(16 x 128-slice) += P(16x64) . V(64 x 128-slice) ----
    v16h ap[2];
#pragma unroll
    for (int c = 0; c < 2; ++c)
      ap[c] = make_frag(*(const v8h*)(prowp + c * 32),
                        *(const v8h*)(prowp + c * 32 + 16));
#pragma unroll
    for (int t = 0; t < 8; ++t) {
      const _Float16* vrow = Vt + (di * D_SLICE + t * 16 + rA) * VT_STRIDE + lh * 8;
#pragma unroll
      for (int c = 0; c < 2; ++c) {
        v16h bb = make_frag(*(const v8h*)(vrow + c * 32),
                            *(const v8h*)(vrow + c * 32 + 16));
        o[t] = __builtin_amdgcn_wmma_f32_16x16x32_f16(false, ap[c], false, bb,
                                                      (short)0, o[t], false, false);
      }
    }
    __syncthreads();   // (D) LDS reads done before next staging
  }

  // ---- epilogue: normalize + post-softmax scale + residual ----
  float linv[8];
#pragma unroll
  for (int v = 0; v < 8; ++v) linv[v] = NORM_FACT / rowl[qi * 16 + v + 8 * lh];
#pragma unroll
  for (int t = 0; t < 8; ++t) {
    int dc = di * D_SLICE + t * 16 + rA;
#pragma unroll
    for (int v = 0; v < 8; ++v) {
      int row = q0 + qi * 16 + v + 8 * lh;
      size_t idx = (size_t)row * D_SZ + dc;
      obase[idx] = o[t][v] * linv[v] + kbase[idx];
    }
  }
}

extern "C" void kernel_launch(void* const* d_in, const int* in_sizes, int n_in,
                              void* d_out, int out_size, void* d_ws, size_t ws_size,
                              hipStream_t stream) {
  const float* key   = (const float*)d_in[0];
  const float* query = (const float*)d_in[1];
  const float* value = (const float*)d_in[2];
  float* out = (float*)d_out;
  (void)in_sizes; (void)n_in; (void)out_size;

  const size_t elems = (size_t)B_SZ * N_SZ * D_SZ;
  const size_t need  = 2 * elems * sizeof(_Float16);   // Kh + Vth = 64 MB
  dim3 grid(N_SZ / M_Q, B_SZ);

  if (ws_size >= need) {
    _Float16* Kh  = (_Float16*)d_ws;
    _Float16* Vth = Kh + elems;
    int n4 = (int)(elems / 4);
    cvt_f16_kernel<<<(n4 + 255) / 256, 256, 0, stream>>>(key, Kh, n4);
    dim3 tg(N_SZ / 32, D_SZ / 32, B_SZ);
    transpose_f16_kernel<<<tg, dim3(32, 8), 0, stream>>>(value, Vth);
    (void)hipFuncSetAttribute((const void*)flash_cross_attn<true>,
                              hipFuncAttributeMaxDynamicSharedMemorySize,
                              (int)SMEM_BYTES);
    flash_cross_attn<true><<<grid, NTHREADS, SMEM_BYTES, stream>>>(
        key, query, value, Kh, Vth, out);
  } else {
    (void)hipFuncSetAttribute((const void*)flash_cross_attn<false>,
                              hipFuncAttributeMaxDynamicSharedMemorySize,
                              (int)SMEM_BYTES);
    flash_cross_attn<false><<<grid, NTHREADS, SMEM_BYTES, stream>>>(
        key, query, value, (const _Float16*)nullptr, (const _Float16*)nullptr, out);
  }
}
